// ViTBlock_34291018891649
// MI455X (gfx1250) — matrix operand order repaired and checked by reading, not compiled, not run on hardware
//
#include <hip/hip_runtime.h>
#include <hip/hip_bf16.h>

typedef __attribute__((ext_vector_type(16))) __bf16 v16bf;
typedef __attribute__((ext_vector_type(8)))  __bf16 v8bf;
typedef __attribute__((ext_vector_type(8)))  float  v8f;
typedef __attribute__((ext_vector_type(4)))  int    v4i;

#define B_    32
#define N_    577
#define C_    768
#define H_    12
#define D_    64
#define HID_  3072
#define M_REAL (B_ * N_)   // 18464
#define M_PAD  18560       // 145 * 128
#define NP_    608         // padded seq len for attention (38*16)
#define BH_    (B_ * H_)   // 384

// ---------------------------------------------------------------------------
// CDNA5 async global->LDS copy (ASYNCcnt-tracked), with safe fallback.
// Probe-discovered signature: (v4i AS1* src, v4i AS3* dst, imm off, imm cpol).
// ---------------------------------------------------------------------------
#if defined(__has_builtin)
#if __has_builtin(__builtin_amdgcn_global_load_async_to_lds_b128)
#define USE_ASYNC_LDS 1
#endif
#endif

typedef __attribute__((address_space(1))) v4i* v4i_gp;
typedef __attribute__((address_space(3))) v4i* v4i_lp;

__device__ __forceinline__ void cp16(const __bf16* g, __bf16* l) {
#if defined(USE_ASYNC_LDS)
  __builtin_amdgcn_global_load_async_to_lds_b128(
      (v4i_gp)(__bf16*)g, (v4i_lp)l, /*offset=*/0, /*cpol=*/0);
#else
  *(v8bf*)l = *(const v8bf*)g;
#endif
}

template <int CNT>
__device__ __forceinline__ void async_wait() {
#if defined(USE_ASYNC_LDS)
#if __has_builtin(__builtin_amdgcn_s_wait_asynccnt)
  __builtin_amdgcn_s_wait_asynccnt(CNT);
#else
  asm volatile("s_wait_asynccnt %0" ::"i"(CNT) : "memory");
#endif
#endif
}

// ---------------------------------------------------------------------------
// WMMA helpers (CDNA5 16x16x32 bf16, f32 accumulate)
// ---------------------------------------------------------------------------
__device__ __forceinline__ v8f wmma_bf16(v16bf a, v16bf b, v8f c) {
  return __builtin_amdgcn_wmma_f32_16x16x32_bf16(
      /*neg_a=*/false, a, /*neg_b=*/false, b,
      /*c_mod=*/(short)0, c, /*reuse_a=*/false, /*reuse_b=*/false);
}

// A fragment: 16x32 (MxK). lane l: m = l&15, half = l>>4.
// elems 0..7  -> K = 8*half .. 8*half+7   (contiguous 16 B)
// elems 8..15 -> K = 16+8*half .. +7      (contiguous 16 B)
__device__ __forceinline__ v16bf load_afrag(const __bf16* base, int stride) {
  int l = threadIdx.x & 31;
  int m = l & 15, hl = l >> 4;
  const __bf16* r = base + m * stride + hl * 8;
  v8bf lo = *(const v8bf*)(r);
  v8bf hi = *(const v8bf*)(r + 16);
  v16bf a;
#pragma unroll
  for (int i = 0; i < 8; ++i) { a[i] = lo[i]; a[8 + i] = hi[i]; }
  return a;
}

// B fragment: 32x16 (KxN). lane l: n = l&15, half = l>>4; elems 0..7 -> K = 8*half+i,
// elems 8..15 -> K = 16+8*half+i (strided by row).
__device__ __forceinline__ v16bf load_bfrag(const __bf16* base, int stride) {
  int l = threadIdx.x & 31;
  int n = l & 15, hl = l >> 4;
  const __bf16* r = base + hl * 8 * stride + n;
  v16bf b;
#pragma unroll
  for (int i = 0; i < 8; ++i) { b[i] = r[i * stride]; b[8 + i] = r[(16 + i) * stride]; }
  return b;
}

// ---------------------------------------------------------------------------
// Tiled bf16 GEMM: C[M_PAD,N] = A[M_PAD,K] @ B[K,N], templated epilogue.
// Block tile 128x64, 4 waves, wave = 32 rows x 64 cols (2x4 accum frags).
// Double-buffered LDS staging via async global->LDS copies (6 chunks/thread
// per tile: 4 for A 128x32, 2 for B 32x64).
// M_PAD % 128 == 0, N % 64 == 0, K % 32 == 0  (guaranteed by padding).
// ---------------------------------------------------------------------------
enum { EP_QKV = 0, EP_RES = 1, EP_GELU = 2 };

__device__ __forceinline__ void stage_tile(
    const __bf16* __restrict__ A, const __bf16* __restrict__ Bm,
    __bf16* sA, __bf16* sB, int tid, int bm, int bn, int N, int K, int k0) {
#pragma unroll
  for (int i = 0; i < 4; ++i) {
    int c = tid + i * 128;
    int row = c >> 2, col = (c & 3) * 8;
    cp16(A + (size_t)(bm * 128 + row) * K + k0 + col, sA + row * 40 + col);
  }
#pragma unroll
  for (int i = 0; i < 2; ++i) {
    int c = tid + i * 128;
    int row = c >> 3, col = (c & 7) * 8;
    cp16(Bm + (size_t)(k0 + row) * N + bn * 64 + col, sB + row * 72 + col);
  }
}

template <int EPI>
__global__ __launch_bounds__(128) void gemm_bf16_k(
    const __bf16* __restrict__ A, const __bf16* __restrict__ Bm,
    int N, int K,
    const float* __restrict__ bias, const float* __restrict__ resid,
    float* __restrict__ outF, __bf16* __restrict__ outB) {
  __shared__ __attribute__((aligned(16))) __bf16 sA[2][128 * 40];  // stride 40
  __shared__ __attribute__((aligned(16))) __bf16 sB[2][32 * 72];   // stride 72
  const int tid  = threadIdx.x;
  const int wv   = tid >> 5;
  const int lane = tid & 31;
  const int bm   = blockIdx.x;
  const int bn   = blockIdx.y;

  v8f acc[2][4] = {};

  const int nk = K >> 5;
  stage_tile(A, Bm, sA[0], sB[0], tid, bm, bn, N, K, 0);

  for (int ks = 0; ks < nk; ++ks) {
    const int cur = ks & 1;
    if (ks + 1 < nk) {
      // prefetch next K-tile into the other buffer (free since ks-1's barrier)
      stage_tile(A, Bm, sA[cur ^ 1], sB[cur ^ 1], tid, bm, bn, N, K,
                 (ks + 1) << 5);
      async_wait<6>();   // async loads retire in order: current tile landed
    } else {
      async_wait<0>();
    }
    __syncthreads();     // all waves' copies for tile ks visible

    v16bf a0 = load_afrag(sA[cur] + (wv * 32 + 0)  * 40, 40);
    v16bf a1 = load_afrag(sA[cur] + (wv * 32 + 16) * 40, 40);
#pragma unroll
    for (int nf = 0; nf < 4; ++nf) {
      v16bf b = load_bfrag(sB[cur] + nf * 16, 72);
      acc[0][nf] = wmma_bf16(a0, b, acc[0][nf]);
      acc[1][nf] = wmma_bf16(a1, b, acc[1][nf]);
    }
    __syncthreads();     // tile ks fully consumed -> buffer reusable
  }

  // epilogue: C/D frag elem r at lane l -> m = r + 8*(l>>4), n = l&15
  const int hl = lane >> 4;
  const int nl = lane & 15;
#pragma unroll
  for (int mf = 0; mf < 2; ++mf)
#pragma unroll
    for (int nf = 0; nf < 4; ++nf)
#pragma unroll
      for (int r = 0; r < 8; ++r) {
        int m = bm * 128 + wv * 32 + mf * 16 + hl * 8 + r;
        int n = bn * 64 + nf * 16 + nl;
        float v = acc[mf][nf][r] + bias[n];
        if constexpr (EPI == EP_QKV) {
          outB[(size_t)m * N + n] = (__bf16)v;
        } else if constexpr (EPI == EP_GELU) {
          float g = 0.5f * v * (1.0f + erff(v * 0.70710678118f));  // exact GELU
          outB[(size_t)m * N + n] = (__bf16)g;
        } else {  // EP_RES: f32 out = acc + bias + residual, guard real rows
          if (m < M_REAL) {
            size_t idx = (size_t)m * N + n;
            outF[idx] = v + resid[idx];
          }
        }
      }
}

// ---------------------------------------------------------------------------
// Attention: one block per (query-tile of 16, b*h). 128 threads (4 waves).
// Q pre-scaled by 1/sqrt(D). Kt is [BH][64][NP], V is [BH][NP][64] (padded,
// zero-filled).  Scores in LDS f32, exact softmax, probs bf16, then P @ V.
// ---------------------------------------------------------------------------
__global__ __launch_bounds__(128) void attn_k(
    const __bf16* __restrict__ Q, const __bf16* __restrict__ Kt,
    const __bf16* __restrict__ V, __bf16* __restrict__ O) {
  __shared__ __attribute__((aligned(16))) __bf16 sQ[16 * 72];
  __shared__ __attribute__((aligned(16))) float  sS[16 * 640];
  __shared__ __attribute__((aligned(16))) __bf16 sP[16 * 640];
  const int qt   = blockIdx.x;   // 0..37
  const int bh   = blockIdx.y;   // 0..383
  const int tid  = threadIdx.x;
  const int wv   = tid >> 5;
  const int lane = tid & 31;
  const int hl   = lane >> 4;
  const int nl   = lane & 15;

  // stage Q tile 16x64 (one 16 B chunk per thread, async)
  {
    int row = tid >> 3, col = (tid & 7) * 8;
    cp16(Q + ((size_t)bh * NP_ + qt * 16 + row) * 64 + col,
         sQ + row * 72 + col);
  }
  async_wait<0>();
  __syncthreads();

  // pass 1: S = Qs @ K^T  (38 col frags split across 4 waves)
  for (int nf = wv; nf < NP_ / 16; nf += 4) {
    v8f acc = {};
#pragma unroll
    for (int kk = 0; kk < 64; kk += 32) {
      v16bf a = load_afrag(sQ + kk, 72);
      v16bf b = load_bfrag(Kt + ((size_t)bh * 64 + kk) * NP_ + nf * 16, NP_);
      acc = wmma_bf16(a, b, acc);
    }
    int ncol = nf * 16 + nl;
    bool ok = ncol < N_;
#pragma unroll
    for (int r = 0; r < 8; ++r)
      sS[(hl * 8 + r) * 640 + ncol] = ok ? acc[r] : -3.0e38f;
  }
  __syncthreads();

  // softmax: 8 threads per row (same-wave subgroup), exact over 577 cols
  {
    int row = tid >> 3, t8 = tid & 7;
    float*  srow = sS + row * 640;
    __bf16* prow = sP + row * 640;
    float mx = -3.0e38f;
    for (int c = t8; c < NP_; c += 8) mx = fmaxf(mx, srow[c]);
#pragma unroll
    for (int o = 4; o >= 1; o >>= 1) mx = fmaxf(mx, __shfl_xor(mx, o, 8));
    float sum = 0.0f;
    for (int c = t8; c < NP_; c += 8) {
      float e = __expf(srow[c] - mx);
      srow[c] = e;
      sum += e;
    }
#pragma unroll
    for (int o = 4; o >= 1; o >>= 1) sum += __shfl_xor(sum, o, 8);
    float inv = 1.0f / sum;
    for (int c = t8; c < NP_; c += 8) prow[c] = (__bf16)(srow[c] * inv);
  }
  __syncthreads();

  // pass 2: O = P @ V  (wave -> 16-wide d frag; padded prob cols are 0)
  {
    int nf = wv;  // 0..3 -> d frags covering D=64
    v8f acc = {};
    for (int kk = 0; kk < NP_; kk += 32) {
      v16bf a = load_afrag(sP + kk, 640);
      v16bf b = load_bfrag(V + ((size_t)bh * NP_ + kk) * 64 + nf * 16, 64);
      acc = wmma_bf16(a, b, acc);
    }
    int d  = nf * 16 + nl;
    int b0 = bh / H_, h = bh % H_;
#pragma unroll
    for (int r = 0; r < 8; ++r) {
      int q = qt * 16 + hl * 8 + r;
      if (q < N_)
        O[((size_t)(b0 * N_ + q)) * C_ + h * 64 + d] = (__bf16)acc[r];
    }
  }
}

// ---------------------------------------------------------------------------
// LayerNorm over C=768, one block (256 thr) per row, bf16 output.
// Rows >= M_REAL write zeros (padding for GEMM A operands).
// ---------------------------------------------------------------------------
__global__ __launch_bounds__(256) void ln_k(
    const float* __restrict__ x, const float* __restrict__ g,
    const float* __restrict__ bta, __bf16* __restrict__ out) {
  const int row = blockIdx.x;
  const int tid = threadIdx.x;
  __bf16* orow = out + (size_t)row * C_;
  if (row >= M_REAL) {
    for (int i = tid; i < C_; i += 256) orow[i] = (__bf16)0.0f;
    return;
  }
  const float* xr = x + (size_t)row * C_;
  float v0 = xr[tid], v1 = xr[tid + 256], v2 = xr[tid + 512];
  __shared__ float red[8];
  __shared__ float smu, srstd;
  const int lane = tid & 31, wv = tid >> 5;

  float s = v0 + v1 + v2;
#pragma unroll
  for (int o = 16; o >= 1; o >>= 1) s += __shfl_xor(s, o, 32);
  if (lane == 0) red[wv] = s;
  __syncthreads();
  if (tid == 0) {
    float t = 0.f;
#pragma unroll
    for (int i = 0; i < 8; ++i) t += red[i];
    smu = t * (1.0f / C_);
  }
  __syncthreads();
  float mu = smu;
  float d0 = v0 - mu, d1 = v1 - mu, d2 = v2 - mu;
  float vs = d0 * d0 + d1 * d1 + d2 * d2;
#pragma unroll
  for (int o = 16; o >= 1; o >>= 1) vs += __shfl_xor(vs, o, 32);
  if (lane == 0) red[wv] = vs;
  __syncthreads();
  if (tid == 0) {
    float t = 0.f;
#pragma unroll
    for (int i = 0; i < 8; ++i) t += red[i];
    srstd = rsqrtf(t * (1.0f / C_) + 1e-5f);
  }
  __syncthreads();
  float rs = srstd;
  orow[tid]       = (__bf16)(d0 * rs * g[tid]       + bta[tid]);
  orow[tid + 256] = (__bf16)(d1 * rs * g[tid + 256] + bta[tid + 256]);
  orow[tid + 512] = (__bf16)(d2 * rs * g[tid + 512] + bta[tid + 512]);
}

// ---------------------------------------------------------------------------
// QKV reshape: [M,2304] bf16 -> per-head Q (scaled), K^T, V padded buffers.
// ---------------------------------------------------------------------------
__global__ __launch_bounds__(256) void qkv_reshape_k(
    const __bf16* __restrict__ qkv, __bf16* __restrict__ Q,
    __bf16* __restrict__ Kt, __bf16* __restrict__ V) {
  int i = blockIdx.x * 256 + threadIdx.x;
  const int total = B_ * H_ * N_ * D_;
  if (i >= total) return;
  int d  = i & 63;
  int n  = (i >> 6) % N_;
  int hb = i / (64 * N_);
  int h = hb % H_, b = hb / H_;
  size_t m = (size_t)b * N_ + n;
  const __bf16* base = qkv + m * (3 * C_);
  int c  = h * 64 + d;
  int bh = b * H_ + h;
  float q = (float)base[c] * 0.125f;  // 1/sqrt(64)
  Q[((size_t)bh * NP_ + n) * 64 + d]  = (__bf16)q;
  Kt[((size_t)bh * 64 + d) * NP_ + n] = base[C_ + c];
  V[((size_t)bh * NP_ + n) * 64 + d]  = base[2 * C_ + c];
}

__global__ __launch_bounds__(256) void f2bf_k(
    const float* __restrict__ in, __bf16* __restrict__ out, int n) {
  int i = blockIdx.x * 256 + threadIdx.x;
  if (i < n) out[i] = (__bf16)in[i];
}

__global__ __launch_bounds__(256) void zero16_k(float4* __restrict__ p, int n16) {
  int i = blockIdx.x * 256 + threadIdx.x;
  if (i < n16) p[i] = make_float4(0.f, 0.f, 0.f, 0.f);
}

// ---------------------------------------------------------------------------
// Orchestration
// ---------------------------------------------------------------------------
extern "C" void kernel_launch(void* const* d_in, const int* in_sizes, int n_in,
                              void* d_out, int out_size, void* d_ws, size_t ws_size,
                              hipStream_t stream) {
  (void)in_sizes; (void)n_in; (void)out_size; (void)ws_size;
  const float* x      = (const float*)d_in[0];
  const float* ln1_g  = (const float*)d_in[1];
  const float* ln1_b  = (const float*)d_in[2];
  const float* qkv_w  = (const float*)d_in[3];
  const float* qkv_b  = (const float*)d_in[4];
  const float* proj_w = (const float*)d_in[5];
  const float* proj_b = (const float*)d_in[6];
  const float* ln2_g  = (const float*)d_in[7];
  const float* ln2_b  = (const float*)d_in[8];
  const float* fc1_w  = (const float*)d_in[9];
  const float* fc1_b  = (const float*)d_in[10];
  const float* fc2_w  = (const float*)d_in[11];
  const float* fc2_b  = (const float*)d_in[12];
  float* out = (float*)d_out;

  char* w = (char*)d_ws;
  size_t off = 0;
  auto take = [&](size_t bytes) -> char* {
    char* p = w + off;
    off += (bytes + 255) & ~(size_t)255;
    return p;
  };

  __bf16* wqkv  = (__bf16*)take((size_t)C_ * 3 * C_ * 2);
  __bf16* wproj = (__bf16*)take((size_t)C_ * C_ * 2);
  __bf16* wfc1  = (__bf16*)take((size_t)C_ * HID_ * 2);
  __bf16* wfc2  = (__bf16*)take((size_t)HID_ * C_ * 2);
  __bf16* hbf   = (__bf16*)take((size_t)M_PAD * C_ * 2);
  __bf16* qkvbf = (__bf16*)take((size_t)M_PAD * 3 * C_ * 2);
  __bf16* Qb    = (__bf16*)take((size_t)BH_ * NP_ * 64 * 2);   // contiguous
  __bf16* Ktb   = (__bf16*)take((size_t)BH_ * 64 * NP_ * 2);   //   zero
  __bf16* Vb    = (__bf16*)take((size_t)BH_ * NP_ * 64 * 2);   //   region
  __bf16* Obf   = (__bf16*)take((size_t)M_PAD * C_ * 2);       //   (Qb..Obf)
  float*  x1    = (float*) take((size_t)M_PAD * C_ * 4);
  __bf16* h2bf  = (__bf16*)take((size_t)M_PAD * C_ * 2);
  __bf16* h3bf  = (__bf16*)take((size_t)M_PAD * HID_ * 2);

  // 1. weights -> bf16
  f2bf_k<<<(C_ * 3 * C_ + 255) / 256, 256, 0, stream>>>(qkv_w, wqkv, C_ * 3 * C_);
  f2bf_k<<<(C_ * C_ + 255) / 256, 256, 0, stream>>>(proj_w, wproj, C_ * C_);
  f2bf_k<<<(C_ * HID_ + 255) / 256, 256, 0, stream>>>(fc1_w, wfc1, C_ * HID_);
  f2bf_k<<<(HID_ * C_ + 255) / 256, 256, 0, stream>>>(fc2_w, wfc2, HID_ * C_);

  // 2. zero the padded Q/Kt/V/O region (contiguous, 16B multiple)
  {
    size_t zb = (size_t)BH_ * NP_ * 64 * 2 * 3 + (size_t)M_PAD * C_ * 2;
    int n16 = (int)(zb / 16);
    zero16_k<<<(n16 + 255) / 256, 256, 0, stream>>>((float4*)Qb, n16);
  }

  // 3. LN1 -> bf16
  ln_k<<<M_PAD, 256, 0, stream>>>(x, ln1_g, ln1_b, hbf);

  // 4. QKV GEMM (+bias, bf16 out)
  gemm_bf16_k<EP_QKV><<<dim3(M_PAD / 128, 3 * C_ / 64), 128, 0, stream>>>(
      hbf, wqkv, 3 * C_, C_, qkv_b, nullptr, nullptr, qkvbf);

  // 5. split/scale/transpose into per-head buffers
  qkv_reshape_k<<<(B_ * H_ * N_ * D_ + 255) / 256, 256, 0, stream>>>(
      qkvbf, Qb, Ktb, Vb);

  // 6. attention (softmax(QK^T) V)
  attn_k<<<dim3(NP_ / 16, BH_), 128, 0, stream>>>(Qb, Ktb, Vb, Obf);

  // 7. proj GEMM + bias + residual(x) -> x1 (f32)
  gemm_bf16_k<EP_RES><<<dim3(M_PAD / 128, C_ / 64), 128, 0, stream>>>(
      Obf, wproj, C_, C_, proj_b, x, x1, nullptr);

  // 8. LN2 -> bf16
  ln_k<<<M_PAD, 256, 0, stream>>>(x1, ln2_g, ln2_b, h2bf);

  // 9. FC1 GEMM + bias + exact GELU -> bf16
  gemm_bf16_k<EP_GELU><<<dim3(M_PAD / 128, HID_ / 64), 128, 0, stream>>>(
      h2bf, wfc1, HID_, C_, fc1_b, nullptr, nullptr, h3bf);

  // 10. FC2 GEMM + bias + residual(x1) -> d_out (f32)
  gemm_bf16_k<EP_RES><<<dim3(M_PAD / 128, C_ / 64), 128, 0, stream>>>(
      h3bf, wfc2, C_, HID_, fc2_b, x1, out, nullptr);
}
